// SSD300_10041633538736
// MI455X (gfx1250) — compile-verified, matrix operand order checked by reading, not run on hardware
//
#include <hip/hip_runtime.h>

// SSD300 post-process for MI455X (gfx1250, wave32).
//
// Kernel 1: TDM (tensor_load_to_lds) stages an 81x128 fp32 label tile into LDS
//   (labels = 181 MB total => memory bound, ~8us floor @ 23.3 TB/s; the TDM
//   streams whole class rows, overlapping across resident blocks), then 8
//   waves compute softmax from LDS. The softmax denominator is reduced on the
//   matrix pipe via v_wmma_f32_16x16x32_f16 (ones-matrix reduction; 16-bit
//   WMMA co-executes with VALU on CDNA5). Candidates (prob > 0.4; at most 2
//   per anchor since softmax sums to 1) are appended as packed 64-bit keys.
//
// Kernel 2: per-image LDS bitonic sort of (score,idx) keys, box decode,
//   greedy NMS (reference semantics), output.

#define NANCH   8732
#define NCLS    81
#define KTOP    400
#define THRV    0.4f
#define IOU_THR 0.7f
#define MAXC    4096
#define SC_XY   0.1f
#define SC_WH   0.2f
#define TILE_A  128          // anchors per block / per TDM tile

typedef __attribute__((ext_vector_type(16))) _Float16 v16h;
typedef __attribute__((ext_vector_type(8)))  float    v8f;
typedef __attribute__((ext_vector_type(4)))  unsigned u32x4;
typedef __attribute__((ext_vector_type(8)))  unsigned u32x8;

__global__ __launch_bounds__(256) void ssd_softmax_collect(
    const float* __restrict__ labels,
    int* __restrict__ cnt, unsigned long long* __restrict__ keys)
{
  __shared__ float tile[NCLS * TILE_A];       // 41,472 B of 320 KB/WGP

  const int b    = blockIdx.y;
  const int tid  = threadIdx.x;
  const int lane = tid & 31;          // wave32
  const int wave = tid >> 5;          // 8 waves / block
  const int a    = lane & 15;         // anchor slot within wave tile
  const int half = lane >> 4;         // class-slot half
  const int n0   = blockIdx.x * TILE_A;
  const int x    = wave * 16 + a;     // anchor within block tile
  const int n    = n0 + x;
  const float* lab = labels + (size_t)b * NCLS * NANCH;

  // ---- TDM: DMA the 2D tile labels[b][0:81][n0:n0+128] into LDS. ----------
  // D# per CDNA5 ISA (2D: groups 0+1 only).
  if (tid < 32) {                     // one wave issues the descriptor
    const unsigned long long ga =
        (unsigned long long)(const void*)(lab + n0);       // tile start (bytes)
    const unsigned ldsOff = (unsigned)(unsigned long long)(void*)&tile[0];
    const unsigned dim0   = (unsigned)(NANCH - n0);        // OOB => zero-fill

    u32x4 g0;
    g0[0] = 1u;                                            // count=1 (valid)
    g0[1] = ldsOff;                                        // lds_addr
    g0[2] = (unsigned)ga;                                  // global_addr[31:0]
    g0[3] = (unsigned)((ga >> 32) & 0x1FFFFFFu)            // global_addr[56:32]
            | 0x80000000u;                                 // type=2 ("image")

    u32x8 g1;
    g1[0] = 0x20000u;                                      // data_size=2 (4B)
    g1[1] = (dim0 & 0xFFFFu) << 16;                        // tensor_dim0[15:0]
    g1[2] = (dim0 >> 16) | ((unsigned)NCLS << 16);         // dim0[31:16]|dim1 lo
    g1[3] = ((unsigned)TILE_A << 16);                      // dim1 hi=0 | tile_dim0
    g1[4] = (unsigned)NCLS;                                // tile_dim1 (tile_dim2=0)
    g1[5] = (unsigned)NANCH;                               // tensor_dim0_stride lo
    g1[6] = 0u;                                            // stride hi | dim1_stride lo
    g1[7] = 0u;

    asm volatile("tensor_load_to_lds %0, %1" :: "s"(g0), "s"(g1) : "memory");
    __builtin_amdgcn_s_wait_tensorcnt(0);
  }

  // Warm the next tile's first rows into L2 (speculative, clamped address).
  {
    const int pn = (n0 + TILE_A < NANCH) ? (n0 + TILE_A) : (NANCH - 1);
    __builtin_prefetch(&lab[(size_t)(tid & 63) * NANCH + pn], 0, 2);
  }
  __syncthreads();

  // ---- Pass A: read 81 logits for this anchor from LDS, running max. ------
  float v[42];
  float mx = -3.0e38f;
#pragma unroll
  for (int q = 0; q < 21; ++q) {
    const int c0 = 4 * q + 2 * half;
    const float x0 = (c0     < NCLS) ? tile[c0       * TILE_A + x] : -3.0e38f;
    const float x1 = (c0 + 1 < NCLS) ? tile[(c0 + 1) * TILE_A + x] : -3.0e38f;
    v[2 * q]     = x0;
    v[2 * q + 1] = x1;
    mx = fmaxf(mx, fmaxf(x0, x1));
  }
  const float amax = fmaxf(mx, __shfl_xor(mx, 16, 32));

  // ---- Pass B: exp in registers, per-lane partial sums. -------------------
  float s0 = 0.0f, s1 = 0.0f;
#pragma unroll
  for (int q = 0; q < 21; ++q) {
    v[2 * q]     = __expf(v[2 * q]     - amax);
    v[2 * q + 1] = __expf(v[2 * q + 1] - amax);
    s0 += v[2 * q];
    s1 += v[2 * q + 1];
  }
  const float s0x = __shfl_xor(s0, 16, 32);
  const float s1x = __shfl_xor(s1, 16, 32);

  // ---- WMMA ones-reduction: A[a][0..3] = anchor a's 4 partials, B = ones =>
  // D[m][n] = total exp-sum of anchor m replicated over all columns. --------
  v16h A, Bn;
#pragma unroll
  for (int i = 0; i < 16; ++i) { A[i] = (_Float16)0.0f; Bn[i] = (_Float16)1.0f; }
  const _Float16 z = (_Float16)0.0f;
  A[0] = half ? z : (_Float16)s0;
  A[1] = half ? z : (_Float16)s1;
  A[2] = half ? z : (_Float16)s0x;
  A[3] = half ? z : (_Float16)s1x;
  v8f Cm = {};
  v8f D = __builtin_amdgcn_wmma_f32_16x16x32_f16(
      /*neg_a=*/false, A, /*neg_b=*/false, Bn,
      /*c_mod=*/(short)0, Cm, /*reuse_a=*/false, /*reuse_b=*/false);

  // Extract row 'a': lane L, VGPR j holds M = j + 8*(L>>4), N = L&15.
  const int jsel = a & 7;
  float t = D[0];
#pragma unroll
  for (int jj = 1; jj < 8; ++jj) t = (jsel == jj) ? D[jj] : t;
  const float u = __shfl_xor(t, 16, 32);
  const float denom = (half == (a >> 3)) ? t : u;
  const float inv = __builtin_amdgcn_rcpf(denom);   // v_rcp_f32, not IEEE div

  // ---- Candidate extraction (sparse: <=2 classes above 0.4 per anchor). ---
  if (n < NANCH) {
#pragma unroll
    for (int q = 0; q < 21; ++q) {
#pragma unroll
      for (int kk = 0; kk < 2; ++kk) {
        const int c = 4 * q + 2 * half + kk;
        if (c >= 1 && c < NCLS) {
          const float p = v[2 * q + kk] * inv;
          if (p > THRV) {
            const int idx = n * NCLS + c;            // flat (N,C) index
            const int pos = atomicAdd(&cnt[b], 1);
            if (pos < MAXC) {
              // descending score, ascending idx tie-break (jax top_k order)
              const unsigned long long key =
                  ((unsigned long long)__float_as_uint(p) << 32) |
                  (unsigned)(0xFFFFFFFFu - (unsigned)idx);
              keys[(size_t)b * MAXC + pos] = key;
            }
          }
        }
      }
    }
  }
}

__global__ __launch_bounds__(512) void ssd_sort_nms(
    const float* __restrict__ locs, const float* __restrict__ dboxes,
    const int* __restrict__ cnt, const unsigned long long* __restrict__ keys,
    float* __restrict__ out, int B)
{
  __shared__ unsigned long long skey[MAXC];   // 32 KB
  __shared__ float s_ob[KTOP * 4];
  __shared__ float s_area[KTOP];
  __shared__ float s_sc[KTOP];
  __shared__ int   s_cls[KTOP];
  __shared__ int   s_keep[KTOP];

  const int b = blockIdx.x;
  const int tid = threadIdx.x;
  int m = cnt[b]; if (m > MAXC) m = MAXC;

  for (int i = tid; i < MAXC; i += 512)
    skey[i] = (i < m) ? keys[(size_t)b * MAXC + i] : 0ULL;
  __syncthreads();

  // Bitonic sort, descending.
  for (int k = 2; k <= MAXC; k <<= 1) {
    for (int j = k >> 1; j > 0; j >>= 1) {
      for (int i = tid; i < MAXC; i += 512) {
        const int ij = i ^ j;
        if (ij > i) {
          const unsigned long long xk = skey[i], yk = skey[ij];
          const bool desc = ((i & k) == 0);
          if (desc ? (xk < yk) : (xk > yk)) { skey[i] = yk; skey[ij] = xk; }
        }
      }
      __syncthreads();
    }
  }

  // Decode top-K candidates.
  for (int i = tid; i < KTOP; i += 512) {
    const unsigned long long key = skey[i];
    const float sc = __uint_as_float((unsigned)(key >> 32));
    const int idx = (int)(0xFFFFFFFFu - (unsigned)key);
    const int valid = (sc > THRV) ? 1 : 0;
    float ox0 = 0, oy0 = 0, ox1 = 0, oy1 = 0, area = 0;
    int cls = 0;
    if (valid) {
      const int bi = idx / NCLS;
      cls = idx - bi * NCLS;
      const float* lp = locs + (size_t)b * 4 * NANCH;
      const float lx = lp[bi];
      const float ly = lp[NANCH + bi];
      const float lw = lp[2 * NANCH + bi];
      const float lh = lp[3 * NANCH + bi];
      const float dx = dboxes[bi * 4 + 0], dy = dboxes[bi * 4 + 1];
      const float dw = dboxes[bi * 4 + 2], dh = dboxes[bi * 4 + 3];
      const float cx = SC_XY * lx * dw + dx;
      const float cy = SC_XY * ly * dh + dy;
      const float w  = __expf(SC_WH * lw) * dw;
      const float h  = __expf(SC_WH * lh) * dh;
      const float off = 10.0f * (float)cls;
      ox0 = cx - 0.5f * w + off; oy0 = cy - 0.5f * h + off;
      ox1 = cx + 0.5f * w + off; oy1 = cy + 0.5f * h + off;
      area = (ox1 - ox0) * (oy1 - oy0);
    }
    s_ob[i * 4 + 0] = ox0; s_ob[i * 4 + 1] = oy0;
    s_ob[i * 4 + 2] = ox1; s_ob[i * 4 + 3] = oy1;
    s_area[i] = area; s_sc[i] = sc; s_cls[i] = cls; s_keep[i] = valid;
  }
  __syncthreads();

  // Greedy NMS, sequential over i (matches reference fori_loop semantics).
  for (int i = 0; i < KTOP; ++i) {
    const int ki = s_keep[i];
    if (ki) {
      const float ax0 = s_ob[i * 4 + 0], ay0 = s_ob[i * 4 + 1];
      const float ax1 = s_ob[i * 4 + 2], ay1 = s_ob[i * 4 + 3];
      const float aa = s_area[i];
      for (int jv = tid; jv < KTOP; jv += 512) {
        if (jv > i && s_keep[jv]) {
          const float lt0 = fmaxf(ax0, s_ob[jv * 4 + 0]);
          const float lt1 = fmaxf(ay0, s_ob[jv * 4 + 1]);
          const float rb0 = fminf(ax1, s_ob[jv * 4 + 2]);
          const float rb1 = fminf(ay1, s_ob[jv * 4 + 3]);
          const float w = fmaxf(rb0 - lt0, 0.0f);
          const float h = fmaxf(rb1 - lt1, 0.0f);
          const float inter = w * h;
          const float iou =
              inter * __builtin_amdgcn_rcpf(aa + s_area[jv] - inter + 1e-12f);
          if (iou > IOU_THR) s_keep[jv] = 0;
        }
      }
    }
    __syncthreads();
  }

  // Outputs concatenated flat: cb (B,K,4), sc (B,K), cl (B,K), keep (B,K).
  float* cb = out;
  float* so = out + (size_t)B * KTOP * 4;
  float* cl = so + (size_t)B * KTOP;
  float* kp = cl + (size_t)B * KTOP;
  for (int i = tid; i < KTOP; i += 512) {
    const int ke = s_keep[i];
    const float off = 10.0f * (float)s_cls[i];
    const size_t r = (size_t)b * KTOP + i;
    cb[r * 4 + 0] = ke ? (s_ob[i * 4 + 0] - off) : 0.0f;
    cb[r * 4 + 1] = ke ? (s_ob[i * 4 + 1] - off) : 0.0f;
    cb[r * 4 + 2] = ke ? (s_ob[i * 4 + 2] - off) : 0.0f;
    cb[r * 4 + 3] = ke ? (s_ob[i * 4 + 3] - off) : 0.0f;
    so[r] = ke ? s_sc[i] : 0.0f;
    cl[r] = ke ? (float)s_cls[i] : 0.0f;
    kp[r] = ke ? 1.0f : 0.0f;
  }
}

extern "C" void kernel_launch(void* const* d_in, const int* in_sizes, int n_in,
                              void* d_out, int out_size, void* d_ws, size_t ws_size,
                              hipStream_t stream) {
  const float* locs   = (const float*)d_in[0];   // (B,4,8732) f32
  const float* labels = (const float*)d_in[1];   // (B,81,8732) f32
  const float* dboxes = (const float*)d_in[2];   // (8732,4) f32
  const int B = in_sizes[0] / (4 * NANCH);

  int* cnt = (int*)d_ws;                                        // B counters
  unsigned long long* keys =
      (unsigned long long*)((char*)d_ws + 1024);                // B x MAXC keys

  hipMemsetAsync(cnt, 0, 1024, stream);   // deterministic across graph replays

  dim3 g1((NANCH + TILE_A - 1) / TILE_A, B);
  ssd_softmax_collect<<<g1, 256, 0, stream>>>(labels, cnt, keys);
  ssd_sort_nms<<<B, 512, 0, stream>>>(locs, dboxes, cnt, keys, (float*)d_out, B);
}
